// NeighborGNN_36060545417821
// MI455X (gfx1250) — compile-verified, matrix operand order checked by reading, not run on hardware
//
#include <hip/hip_runtime.h>

typedef __attribute__((ext_vector_type(16))) __bf16 v16bf;
typedef __attribute__((ext_vector_type(8)))  float  v8f;
typedef unsigned short u16;

union FragBF {
    uint4 q[2];
    v16bf v;
};

__device__ __forceinline__ u16 f2bf(float f) {
    unsigned u = __float_as_uint(f);
    // round-to-nearest-even bf16
    unsigned r = (u + 0x7FFFu + ((u >> 16) & 1u)) >> 16;
    return (u16)r;
}

__device__ __forceinline__ float softmax3(const float* __restrict__ sw, int b) {
    float a0 = sw[0], a1 = sw[1], a2 = sw[2];
    float m  = fmaxf(a0, fmaxf(a1, a2));
    float e0 = __expf(a0 - m), e1 = __expf(a1 - m), e2 = __expf(a2 - m);
    float s  = e0 + e1 + e2;
    float e  = (b == 0) ? e0 : ((b == 1) ? e1 : e2);
    return e / s;
}

// ---------------- prep kernels ----------------

__global__ void k_cvt_bf16(const float* __restrict__ src, u16* __restrict__ dst, long n) {
    long t = (long)blockIdx.x * blockDim.x + threadIdx.x;
    if (t < n) dst[t] = f2bf(src[t]);
}

__global__ void k_zero_u16(u16* __restrict__ dst, long n) {
    long t = (long)blockIdx.x * blockDim.x + threadIdx.x;
    if (t < n) dst[t] = 0;
}

// W [K, Nc] row-major f32  ->  Wt [Nc, K] row-major bf16
__global__ void k_wt(const float* __restrict__ W, u16* __restrict__ Wt, int K, int Nc) {
    long t = (long)blockIdx.x * blockDim.x + threadIdx.x;
    if (t >= (long)K * Nc) return;
    int n = (int)(t % Nc);
    int k = (int)(t / Nc);
    Wt[(size_t)n * K + k] = f2bf(W[t]);
}

__global__ void k_deg_init(float* __restrict__ deg, int n) {
    int t = blockIdx.x * blockDim.x + threadIdx.x;
    if (t < n) deg[t] = 1.0f;               // self loop
}

__global__ void k_deg_count(const int* __restrict__ dstv, int E, float* __restrict__ deg) {
    int t = blockIdx.x * blockDim.x + threadIdx.x;
    if (t < E) atomicAdd(&deg[dstv[t]], 1.0f);
}

__global__ void k_rsqrt(float* __restrict__ deg, int n) {
    int t = blockIdx.x * blockDim.x + threadIdx.x;
    if (t < n) {
        float d = deg[t];
        deg[t] = (d > 0.0f) ? rsqrtf(d) : 0.0f;
    }
}

// ---------------- WMMA GEMM ----------------
// C[M, Nout] f32 = A[M, K] bf16 (row-major) x Bt[Nout, K] bf16 (row-major, i.e. B K-major)
// One wave computes a 64x64 tile as 4x4 grid of 16x16 WMMA sub-tiles:
//  - 4 A + 4 B fragments per K-step feed 16 WMMAs (~32 FLOP/byte of L2 traffic)
//  - row displacements are offsets mi*16*K, nj*16*K from one base (K is a runtime argument)
//  - __launch_bounds__(256,1) lifts the VGPR budget so the 128 accumulator
//    VGPRs allocate without scratch spills
// M multiple of 64 (caller pads activations), K multiple of 32, Nout multiple of 64.
#define MT 4
#define NT 4
__global__ void __launch_bounds__(256, 1)
k_wmma_gemm(const u16* __restrict__ A, const u16* __restrict__ Bt,
            float* __restrict__ C, int M, int Nout, int K) {
    const int lane  = threadIdx.x & 31;
    const int wave  = (int)(((long)blockIdx.x * blockDim.x + threadIdx.x) >> 5);
    const int ntile = Nout >> 6;
    const int njobs = (M >> 6) * ntile;
    if (wave >= njobs) return;
    const int m0   = (wave / ntile) << 6;
    const int n0   = (wave % ntile) << 6;
    const int half = lane >> 4;                  // 0: K{0..7,16..23}  1: K{8..15,24..31}
    const int lr   = lane & 15;

    v8f acc[MT][NT];
#pragma unroll
    for (int mi = 0; mi < MT; ++mi)
#pragma unroll
        for (int nj = 0; nj < NT; ++nj)
            acc[mi][nj] = (v8f){0.f, 0.f, 0.f, 0.f, 0.f, 0.f, 0.f, 0.f};

    // single per-lane pointer for A and B; sub-tile rows addressed via
    // runtime element offsets (mi*16*K, nj*16*K), K being a kernel argument
    const u16* ap = A  + (size_t)(m0 + lr) * K + half * 8;
    const u16* bp = Bt + (size_t)(n0 + lr) * K + half * 8;

    for (int k0 = 0; k0 < K; k0 += 32, ap += 32, bp += 32) {
        FragBF a[MT];
#pragma unroll
        for (int mi = 0; mi < MT; ++mi) {
            a[mi].q[0] = *reinterpret_cast<const uint4*>(ap + (size_t)mi * 16 * K);
            a[mi].q[1] = *reinterpret_cast<const uint4*>(ap + (size_t)mi * 16 * K + 16);
            __builtin_prefetch(ap + (size_t)mi * 16 * K + 32, 0, 3);
        }
#pragma unroll
        for (int nj = 0; nj < NT; ++nj) {
            FragBF b;
            b.q[0] = *reinterpret_cast<const uint4*>(bp + (size_t)nj * 16 * K);
            b.q[1] = *reinterpret_cast<const uint4*>(bp + (size_t)nj * 16 * K + 16);
#pragma unroll
            for (int mi = 0; mi < MT; ++mi)
                acc[mi][nj] = __builtin_amdgcn_wmma_f32_16x16x32_bf16(
                    false, a[mi].v, false, b.v, (short)0, acc[mi][nj], false, false);
        }
    }

    // C/D layout: VGPR r -> row (sub-tile base) + half*8 + r ; lane%16 -> column
#pragma unroll
    for (int mi = 0; mi < MT; ++mi) {
#pragma unroll
        for (int nj = 0; nj < NT; ++nj) {
            float* cp = C + (size_t)(m0 + mi * 16 + half * 8) * Nout + n0 + nj * 16 + lr;
#pragma unroll
            for (int r = 0; r < 8; ++r)
                cp[(size_t)r * Nout] = acc[mi][nj][r];
        }
    }
}

// ---------------- aggregation kernels ----------------

// agg[i,f] = h[i,f] * dinv[i]^2        (self-loop term, also initializes agg)
__global__ void k_self_init(const float* __restrict__ h, const float* __restrict__ dinv,
                            float* __restrict__ agg, int n, int F) {
    long t = (long)blockIdx.x * blockDim.x + threadIdx.x;
    if (t >= (long)n * F) return;
    int i = (int)(t / F);
    float di = dinv[i];
    agg[t] = h[t] * di * di;
}

// agg[dst] += h[src] * dinv[src]*dinv[dst]   (float4 per thread)
__global__ void k_edge_scatter(const int* __restrict__ src, const int* __restrict__ dstv, int E,
                               const float* __restrict__ dinv, const float* __restrict__ h,
                               float* __restrict__ agg, int F) {
    const int chunks = F >> 2;
    long t = (long)blockIdx.x * blockDim.x + threadIdx.x;
    if (t >= (long)E * chunks) return;
    int e = (int)(t / chunks);
    int c = (int)(t % chunks);
    int s = src[e], d = dstv[e];
    float nrm = dinv[s] * dinv[d];
    const float4 hv = *reinterpret_cast<const float4*>(h + (size_t)s * F + 4 * c);
    float* ap = agg + (size_t)d * F + 4 * c;
    atomicAdd(ap + 0, hv.x * nrm);
    atomicAdd(ap + 1, hv.y * nrm);
    atomicAdd(ap + 2, hv.z * nrm);
    atomicAdd(ap + 3, hv.w * nrm);
}

// hb16[i,f] = bf16( relu( agg[i,f] + bias[f] ) )
__global__ void k_bias_relu_bf(const float* __restrict__ agg, const float* __restrict__ bias,
                               u16* __restrict__ out, int n, int F) {
    long t = (long)blockIdx.x * blockDim.x + threadIdx.x;
    if (t >= (long)n * F) return;
    int f = (int)(t % F);
    float v = agg[t] + bias[f];
    out[t] = f2bf(fmaxf(v, 0.0f));
}

// out[i,f] (=|+=) w_b * ( h[i,f]*dinv[i]^2 + b2[f] )
__global__ void k_out_self(const float* __restrict__ h, const float* __restrict__ dinv,
                           const float* __restrict__ bias, const float* __restrict__ sw,
                           int branch, float* __restrict__ out, int n, int F) {
    long t = (long)blockIdx.x * blockDim.x + threadIdx.x;
    if (t >= (long)n * F) return;
    int i = (int)(t / F);
    int f = (int)(t % F);
    float w  = softmax3(sw, branch);
    float di = dinv[i];
    float v  = w * (h[t] * di * di + bias[f]);
    if (branch == 0) out[t] = v;
    else             out[t] += v;
}

// out[dst] += w_b * norm * h[src]
__global__ void k_out_edge(const int* __restrict__ src, const int* __restrict__ dstv, int E,
                           const float* __restrict__ dinv, const float* __restrict__ h,
                           const float* __restrict__ sw, int branch,
                           float* __restrict__ out, int F) {
    const int chunks = F >> 2;
    long t = (long)blockIdx.x * blockDim.x + threadIdx.x;
    if (t >= (long)E * chunks) return;
    int e = (int)(t / chunks);
    int c = (int)(t % chunks);
    int s = src[e], d = dstv[e];
    float w   = softmax3(sw, branch);
    float nrm = w * dinv[s] * dinv[d];
    const float4 hv = *reinterpret_cast<const float4*>(h + (size_t)s * F + 4 * c);
    float* op = out + (size_t)d * F + 4 * c;
    atomicAdd(op + 0, hv.x * nrm);
    atomicAdd(op + 1, hv.y * nrm);
    atomicAdd(op + 2, hv.z * nrm);
    atomicAdd(op + 3, hv.w * nrm);
}

// ---------------- launcher ----------------

extern "C" void kernel_launch(void* const* d_in, const int* in_sizes, int n_in,
                              void* d_out, int out_size, void* d_ws, size_t ws_size,
                              hipStream_t stream) {
    (void)n_in; (void)out_size; (void)ws_size;

    const int DIN = 256, DH = 768, DOUT = 1536;
    const int N = in_sizes[0] / DIN;
    const int M_pad = (N + 63) & ~63;            // GEMM row padding (64-row tiles)

    const float* x  = (const float*)d_in[0];
    const int*  edge[3] = { (const int*)d_in[1], (const int*)d_in[2], (const int*)d_in[3] };
    const int   E[3]    = { in_sizes[1] / 2, in_sizes[2] / 2, in_sizes[3] / 2 };
    const float* W1[3] = { (const float*)d_in[4],  (const float*)d_in[8],  (const float*)d_in[12] };
    const float* B1[3] = { (const float*)d_in[5],  (const float*)d_in[9],  (const float*)d_in[13] };
    const float* W2[3] = { (const float*)d_in[6],  (const float*)d_in[10], (const float*)d_in[14] };
    const float* B2[3] = { (const float*)d_in[7],  (const float*)d_in[11], (const float*)d_in[15] };
    const float* sw    = (const float*)d_in[16];
    float* out = (float*)d_out;

    // workspace carve-out (256B aligned)
    char* ws = (char*)d_ws;
    size_t off = 0;
    auto carve = [&](size_t bytes) -> char* {
        char* p = ws + off;
        off = (off + bytes + 255) & ~(size_t)255;
        return p;
    };
    float* dinv  = (float*)carve((size_t)N * sizeof(float));
    u16*   xb16  = (u16*)  carve((size_t)M_pad * DIN * sizeof(u16));
    u16*   wt1   = (u16*)  carve((size_t)3 * DH * DIN * sizeof(u16));
    u16*   wt2   = (u16*)  carve((size_t)3 * DOUT * DH * sizeof(u16));
    float* hbuf  = (float*)carve((size_t)M_pad * DOUT * sizeof(float)); // H1 (DH cols) then H2 (DOUT cols)
    float* abuf  = (float*)carve((size_t)N * DH * sizeof(float));
    u16*   hb16  = (u16*)  carve((size_t)M_pad * DH * sizeof(u16));

    const int TB = 256;
    auto cdiv = [](long a, long b) -> long { return (a + b - 1) / b; };

    // ---- prep: x -> bf16 (+ zero pad rows), transpose+convert all weights ----
    k_cvt_bf16<<<cdiv((long)N * DIN, TB), TB, 0, stream>>>(x, xb16, (long)N * DIN);
    if (M_pad > N) {
        k_zero_u16<<<cdiv((long)(M_pad - N) * DIN, TB), TB, 0, stream>>>(
            xb16 + (size_t)N * DIN, (long)(M_pad - N) * DIN);
        k_zero_u16<<<cdiv((long)(M_pad - N) * DH, TB), TB, 0, stream>>>(
            hb16 + (size_t)N * DH, (long)(M_pad - N) * DH);
    }
    for (int b = 0; b < 3; ++b) {
        k_wt<<<cdiv((long)DIN * DH, TB), TB, 0, stream>>>(W1[b], wt1 + (size_t)b * DH * DIN, DIN, DH);
        k_wt<<<cdiv((long)DH * DOUT, TB), TB, 0, stream>>>(W2[b], wt2 + (size_t)b * DOUT * DH, DH, DOUT);
    }

    for (int b = 0; b < 3; ++b) {
        const int* srcv = edge[b];
        const int* dstv = edge[b] + E[b];

        // degrees / dinv (deg over dst + self loop)
        k_deg_init <<<cdiv(N, TB), TB, 0, stream>>>(dinv, N);
        k_deg_count<<<cdiv(E[b], TB), TB, 0, stream>>>(dstv, E[b], dinv);
        k_rsqrt    <<<cdiv(N, TB), TB, 0, stream>>>(dinv, N);

        // layer 1: H1 = x @ W1   (M_pad x DH)
        {
            long waves = (long)(M_pad >> 6) * (DH >> 6);
            k_wmma_gemm<<<cdiv(waves * 32, TB), TB, 0, stream>>>(
                xb16, wt1 + (size_t)b * DH * DIN, hbuf, M_pad, DH, DIN);
        }
        k_self_init   <<<cdiv((long)N * DH, TB), TB, 0, stream>>>(hbuf, dinv, abuf, N, DH);
        k_edge_scatter<<<cdiv((long)E[b] * (DH >> 2), TB), TB, 0, stream>>>(
            srcv, dstv, E[b], dinv, hbuf, abuf, DH);
        k_bias_relu_bf<<<cdiv((long)N * DH, TB), TB, 0, stream>>>(abuf, B1[b], hb16, N, DH);

        // layer 2: H2 = relu_out @ W2   (M_pad x DOUT)
        {
            long waves = (long)(M_pad >> 6) * (DOUT >> 6);
            k_wmma_gemm<<<cdiv(waves * 32, TB), TB, 0, stream>>>(
                hb16, wt2 + (size_t)b * DOUT * DH, hbuf, M_pad, DOUT, DH);
        }
        k_out_self<<<cdiv((long)N * DOUT, TB), TB, 0, stream>>>(
            hbuf, dinv, B2[b], sw, b, out, N, DOUT);
        k_out_edge<<<cdiv((long)E[b] * (DOUT >> 2), TB), TB, 0, stream>>>(
            srcv, dstv, E[b], dinv, hbuf, sw, b, out, DOUT);
    }
}
